// spatio_module_23630910063097
// MI455X (gfx1250) — matrix operand order repaired and checked by reading, not compiled, not run on hardware
//
#include <hip/hip_runtime.h>
#include <hip/hip_bf16.h>

// ---------------------------------------------------------------------------
// Problem constants (from the reference)
// ---------------------------------------------------------------------------
constexpr int Bc  = 16;
constexpr int Nc  = 4096;
constexpr int IN  = 192;   // HORIZON * E
constexpr int TD  = 64;    // transformer dim
constexpr int FF  = 128;   // ffn dim
constexpr int OD  = 192;   // out dim
constexpr int TOK = Bc * Nc;   // 65536 tokens
constexpr int NTILE = TOK / 16;

// ws layout (float units)
constexpr size_t OFF_Q    = 0;                          // TOK*64
constexpr size_t OFF_P    = OFF_Q + (size_t)TOK*TD;     // TOK*64
constexpr size_t OFF_ACC  = OFF_P + (size_t)TOK*TD;     // 16*4*832
constexpr size_t OFF_FKVN = OFF_ACC + (size_t)Bc*4*832; // 16*4*784
constexpr size_t OFF_PAKW = OFF_FKVN + (size_t)Bc*4*784;// packed weights (bf16)
constexpr size_t PAKW_BF16 = 114688;                    // total packed bf16 elems
constexpr size_t OFF_XPAK = OFF_PAKW + PAKW_BF16/2;     // packed x (bf16)
constexpr int ACC_N = Bc*4*832;

// packed-weight sub-offsets (bf16 elements), fragment = 512 bf16 (32 lanes x 16)
constexpr size_t PW_Q    = 0;      // 24 frags
constexpr size_t PW_K    = 12288;  // 24
constexpr size_t PW_V    = 24576;  // 24
constexpr size_t PW_IN   = 36864;  // 24
constexpr size_t PW_1    = 49152;  // 16
constexpr size_t PW_2    = 57344;  // 16
constexpr size_t PW_OUT  = 65536;  // 24
constexpr size_t PW_POOL = 77824;  // 72

// ---------------------------------------------------------------------------
// CDNA5 WMMA helpers (gfx1250, wave32): D = A(16x32 bf16) * B(32x16 bf16) + C(f32)
// ---------------------------------------------------------------------------
typedef __attribute__((ext_vector_type(16))) __bf16 bhalf16;
typedef __attribute__((ext_vector_type(8)))  float  f32x8;

// A fragment from f32 source: row = lane&15; lanes 0-15 hold K{0..7,16..23},
// lanes 16-31 hold K{8..15,24..31}
__device__ inline bhalf16 ld_a16(const float* s, int ld, int lane) {
  const int row = lane & 15;
  const int kb  = (lane & 16) ? 8 : 0;
  bhalf16 a;
#pragma unroll
  for (int i = 0; i < 16; ++i) {
    const int k = ((i < 8) ? 0 : 16) + kb + (i & 7);
    a[i] = (__bf16)s[row * ld + k];
  }
  return a;
}

// pre-packed fragment load: one 32-byte vector load per lane
__device__ inline bhalf16 ld_pk(const __bf16* base, int frag, int lane) {
  return *(const bhalf16*)(base + (size_t)frag * 512 + lane * 16);
}

__device__ inline f32x8 wmma_bf16(bhalf16 a, bhalf16 b, f32x8 c) {
  return __builtin_amdgcn_wmma_f32_16x16x32_bf16(false, a, false, b,
                                                 (short)0, c, false, false);
}

// ---------------------------------------------------------------------------
// Kernel 0: zero the f_kv accumulators
// ---------------------------------------------------------------------------
__global__ void k_zero(float* __restrict__ p, int n) {
  int i = blockIdx.x * blockDim.x + threadIdx.x;
  if (i < n) p[i] = 0.f;
}

// ---------------------------------------------------------------------------
// Kernel P: pack one f32 weight (K x Ncols, row-major) into bf16 B-fragment
// layout. blockIdx = fragment (ks*(Ncols/16)+j), 32 threads = lanes.
// B-frag layout: col = lane&15; lanes 0-15 hold K{0..7,16..23}, lanes 16-31 K{8..15,24..31}.
// ---------------------------------------------------------------------------
__global__ __launch_bounds__(32) void k_packw(const float* __restrict__ W,
                                              int Ncols, __bf16* __restrict__ dst) {
  const int lane = threadIdx.x;
  const int nt = Ncols >> 4;
  const int ks = blockIdx.x / nt;
  const int j  = blockIdx.x % nt;
  const int col = j * 16 + (lane & 15);
  const int kb  = ks * 32 + ((lane & 16) ? 8 : 0);
  bhalf16 v;
#pragma unroll
  for (int i = 0; i < 16; ++i) v[i] = (__bf16)W[(size_t)(kb + ((i < 8) ? 0 : 16) + (i & 7)) * Ncols + col];
  *(bhalf16*)(dst + (size_t)blockIdx.x * 512 + lane * 16) = v;
}

// ---------------------------------------------------------------------------
// Kernel 1: QKV + Win projections (WMMA) + in-LDS linear-attention statistics.
// Also writes the bf16 A-fragment copy of x for reuse by k_tail.
// 256 threads = 8 waves, 1 wave = 16 tokens, block = 128 tokens (same b).
// Per (b,h) accumulator layout (832 floats):
//   [0..767]    f_kv for k=1..3   ((k-1)*256 + f*16 + c)
//   [768..783]  sum_n V[c]        (k=0 numerator)
//   [784..831]  sumN for k=1..3   ((k-1)*16 + f)
// ---------------------------------------------------------------------------
__global__ __launch_bounds__(256) void k_proj_stats(
    const float* __restrict__ x,
    const __bf16* __restrict__ pw,           // packed weights base
    const float* __restrict__ bq, const float* __restrict__ bk,
    const float* __restrict__ bv,
    float* __restrict__ Qws, float* __restrict__ Pws,
    float* __restrict__ acc, __bf16* __restrict__ xpak)
{
  __shared__ float s_acc[4 * 832];      // 3328 floats, block accumulators
  __shared__ float s_kv[8][16][32];     // per-wave K|V tile staging

  const int tid  = threadIdx.x;
  const int lane = tid & 31;
  const int wid  = tid >> 5;
  const int c16  = lane & 15;
  const int hi   = lane >> 4;

  for (int i = tid; i < 3328; i += 256) s_acc[i] = 0.f;
  __syncthreads();

  const int tt = blockIdx.x * 8 + wid;          // global tile index
  const int t0 = tt * 16;                       // global token base of this wave

  bhalf16 xa[6];
#pragma unroll
  for (int ks = 0; ks < 6; ++ks) {
    xa[ks] = ld_a16(x + (size_t)t0 * IN + ks * 32, IN, lane);
    *(bhalf16*)(xpak + ((size_t)tt * 6 + ks) * 512 + lane * 16) = xa[ks];
  }

  // ---- Q and proj (x @ Wq + bq ; x @ Win) ----
#pragma unroll
  for (int jj = 0; jj < 4; ++jj) {
    f32x8 cq = {}; f32x8 cp = {};
#pragma unroll
    for (int ks = 0; ks < 6; ++ks) {
      cq = wmma_bf16(xa[ks], ld_pk(pw + PW_Q,  ks * 4 + jj, lane), cq);
      cp = wmma_bf16(xa[ks], ld_pk(pw + PW_IN, ks * 4 + jj, lane), cp);
    }
    const int col = jj * 16 + c16;
    const float bqv = bq[col];
#pragma unroll
    for (int r = 0; r < 8; ++r) {
      const int row = r + hi * 8;
      Qws[(size_t)(t0 + row) * TD + col] = cq[r] + bqv;
      Pws[(size_t)(t0 + row) * TD + col] = cp[r];
    }
  }

  // ---- per head: K,V tiles + power statistics ----
  for (int h = 0; h < 4; ++h) {
    f32x8 ck = {}; f32x8 cv = {};
#pragma unroll
    for (int ks = 0; ks < 6; ++ks) {
      ck = wmma_bf16(xa[ks], ld_pk(pw + PW_K, ks * 4 + h, lane), ck);
      cv = wmma_bf16(xa[ks], ld_pk(pw + PW_V, ks * 4 + h, lane), cv);
    }
    const float bkv = bk[h * 16 + c16];
    const float bvv = bv[h * 16 + c16];
#pragma unroll
    for (int r = 0; r < 8; ++r) {
      const int row = r + hi * 8;
      s_kv[wid][row][c16]      = ck[r] + bkv;   // K[t][f]
      s_kv[wid][row][16 + c16] = cv[r] + bvv;   // V[t][c]
    }
    __syncthreads();

    // lane handles f = c16, c-range = hi*8 .. hi*8+7
    float a1[8] = {}, a2[8] = {}, a3[8] = {};
    float s1 = 0.f, s2 = 0.f, s3 = 0.f, sv = 0.f;
#pragma unroll
    for (int t = 0; t < 16; ++t) {
      const float kv = s_kv[wid][t][c16];
      const float k2 = kv * kv;
      const float k3 = k2 * kv;
      s1 += kv; s2 += k2; s3 += k3;
      sv += s_kv[wid][t][16 + c16];
#pragma unroll
      for (int r = 0; r < 8; ++r) {
        const float v = s_kv[wid][t][16 + hi * 8 + r];
        a1[r] += kv * v; a2[r] += k2 * v; a3[r] += k3 * v;
      }
    }
    float* base = s_acc + h * 832;
#pragma unroll
    for (int r = 0; r < 8; ++r) {
      const int c = hi * 8 + r;
      atomicAdd(base +   0 + c16 * 16 + c, a1[r]);
      atomicAdd(base + 256 + c16 * 16 + c, a2[r]);
      atomicAdd(base + 512 + c16 * 16 + c, a3[r]);
    }
    if (hi == 0) {
      atomicAdd(base + 784 +  0 + c16, s1);
      atomicAdd(base + 784 + 16 + c16, s2);
      atomicAdd(base + 784 + 32 + c16, s3);
    } else {
      atomicAdd(base + 768 + c16, sv);          // k=0: sum_n V
    }
    __syncthreads();
  }

  // ---- one global flush per block ----
  const int b = blockIdx.x >> 5;                // 32 blocks per batch
  float* g = acc + (size_t)b * 3328;
  for (int i = tid; i < 3328; i += 256) atomicAdd(g + i, s_acc[i]);
}

// ---------------------------------------------------------------------------
// Kernel 2: normalize f_kv and fold in Pi[h,k] and 1/3!  (tiny)
// FKVN per (b,h): [0..15] k=0 vector; [16 + (k-1)*256 + f*16 + c] for k=1..3
// ---------------------------------------------------------------------------
__global__ __launch_bounds__(256) void k_fin(
    const float* __restrict__ acc, const float* __restrict__ Pi,
    float* __restrict__ fkvn)
{
  const int bh = blockIdx.x;        // b*4 + h
  const int h  = bh & 3;
  const float inv6 = 1.0f / 6.0f;
  const float* a = acc  + (size_t)bh * 832;
  float*       o = fkvn + (size_t)bh * 784;
  const int tid = threadIdx.x;
  if (tid < 16)
    o[tid] = a[768 + tid] / ((float)Nc + 1e-8f) * Pi[h * 4 + 0] * inv6;
  for (int i = tid; i < 768; i += 256) {
    const int k = i >> 8;           // 0..2 -> power k+1
    const int f = (i >> 4) & 15;
    const float sn = a[784 + k * 16 + f];
    o[16 + i] = a[i] / (sn + 1e-8f) * 0.25f /* 1/sqrt(16) */
              * Pi[h * 4 + (k + 1)] * inv6;
  }
}

// ---------------------------------------------------------------------------
// Kernel 3: attention gather (raw-view reshape!), LN1, FFN (WMMA), LN2,
//           fused output GEMM  x@weights_pool + res@Wout + bout   (WMMA)
// 128 threads = 4 waves, 1 wave = 16 tokens, block = 64 tokens (same b).
// ---------------------------------------------------------------------------
__global__ __launch_bounds__(128) void k_tail(
    const __bf16* __restrict__ xpak, const float* __restrict__ Qws,
    const float* __restrict__ Pws,   const float* __restrict__ fkvn,
    const __bf16* __restrict__ pw,
    const float* __restrict__ b1,   const float* __restrict__ b2,
    const float* __restrict__ ln_g, const float* __restrict__ ln_b,
    const float* __restrict__ bout, float* __restrict__ out)
{
  __shared__ float s_f[4 * 784];        // fkvn for this batch (all heads)
  __shared__ float s_T[4][16][66];      // per-wave res tile
  __shared__ float s_H[4][16][130];     // per-wave ffn hidden tile

  const int tid  = threadIdx.x;
  const int lane = tid & 31;
  const int wid  = tid >> 5;
  const int c16  = lane & 15;
  const int hi   = lane >> 4;
  const int b    = blockIdx.x >> 6;     // 64 blocks per batch

  for (int i = tid; i < 3136; i += 128) s_f[i] = fkvn[(size_t)b * 3136 + i];
  __syncthreads();

  const int tt    = blockIdx.x * 4 + wid;        // global tile index
  const int t0    = tt * 16;                     // global token base of wave
  const int nloc0 = t0 - b * Nc;                 // token-in-batch base
  const int t     = lane & 15;                   // token within tile
  const int chalf = hi * 32;                     // this lane's column half
  const int nloc  = nloc0 + t;
  const int gtok  = t0 + t;

  // ---- linear attention (faithful to reshape(B,H,N,D)->(B,N,64) raw view) ----
  float r0[32];
#pragma unroll
  for (int qg = 0; qg < 2; ++qg) {
    const int q    = hi * 2 + qg;        // output col group c>>4
    const int idx  = nloc * 4 + q;       // flat/(D) index inside batch
    const int hatt = idx >> 12;          // which head the raw view hits
    const int np   = idx & 4095;         // which token the raw view hits
    const float* Qrow = Qws + (size_t)(b * Nc + np) * TD + hatt * 16;
    const float* fb   = s_f + hatt * 784;
    float av[16];
#pragma unroll
    for (int cc = 0; cc < 16; ++cc) av[cc] = fb[cc];      // k=0 term
#pragma unroll
    for (int f = 0; f < 16; ++f) {
      const float qv = Qrow[f];
      const float q2 = qv * qv, q3 = q2 * qv;
      const float* m1 = fb + 16 + f * 16;
      const float* m2 = m1 + 256;
      const float* m3 = m2 + 256;
#pragma unroll
      for (int cc = 0; cc < 16; ++cc)
        av[cc] += qv * m1[cc] + q2 * m2[cc] + q3 * m3[cc];
    }
#pragma unroll
    for (int cc = 0; cc < 16; ++cc) r0[qg * 16 + cc] = av[cc];
  }
  {
    const float* Prow = Pws + (size_t)gtok * TD + chalf;
#pragma unroll
    for (int i = 0; i < 32; ++i) r0[i] += Prow[i];
  }

  // ---- LN1 (pair of lanes shares a token via shfl_xor 16) ----
  float sum = 0.f, sq = 0.f;
#pragma unroll
  for (int i = 0; i < 32; ++i) { sum += r0[i]; sq += r0[i] * r0[i]; }
  sum += __shfl_xor(sum, 16); sq += __shfl_xor(sq, 16);
  const float mu  = sum * (1.0f / 64.0f);
  const float var = sq * (1.0f / 64.0f) - mu * mu;
  const float ri  = rsqrtf(var + 1e-5f);
#pragma unroll
  for (int i = 0; i < 32; ++i) {
    const int c = chalf + i;
    s_T[wid][t][c] = (r0[i] - mu) * ri * ln_g[c] + ln_b[c];
  }
  __syncthreads();

  // ---- FFN layer 1: res(16x64) @ W1(64x128), relu, to LDS ----
  bhalf16 ra0 = ld_a16(&s_T[wid][0][0],      66, lane);
  bhalf16 ra1 = ld_a16(&s_T[wid][0][0] + 32, 66, lane);
#pragma unroll
  for (int j = 0; j < 8; ++j) {
    f32x8 c_ = {};
    c_ = wmma_bf16(ra0, ld_pk(pw + PW_1, 0 * 8 + j, lane), c_);
    c_ = wmma_bf16(ra1, ld_pk(pw + PW_1, 1 * 8 + j, lane), c_);
    const int col = j * 16 + c16;
    const float bb = b1[col];
#pragma unroll
    for (int r = 0; r < 8; ++r)
      s_H[wid][r + hi * 8][col] = fmaxf(c_[r] + bb, 0.f);
  }
  __syncthreads();

  // ---- FFN layer 2 + residual into s_T ----
  bhalf16 ha[4];
#pragma unroll
  for (int ks = 0; ks < 4; ++ks)
    ha[ks] = ld_a16(&s_H[wid][0][0] + ks * 32, 130, lane);
#pragma unroll
  for (int j = 0; j < 4; ++j) {
    f32x8 c_ = {};
#pragma unroll
    for (int ks = 0; ks < 4; ++ks)
      c_ = wmma_bf16(ha[ks], ld_pk(pw + PW_2, ks * 4 + j, lane), c_);
    const int col = j * 16 + c16;
    const float bb = b2[col];
#pragma unroll
    for (int r = 0; r < 8; ++r) {
      const int row = r + hi * 8;
      s_T[wid][row][col] = s_T[wid][row][col] + c_[r] + bb;
    }
  }
  __syncthreads();

  // ---- LN2 ----
  float v2[32];
  float sum2 = 0.f, sq2 = 0.f;
#pragma unroll
  for (int i = 0; i < 32; ++i) {
    const float v = s_T[wid][t][chalf + i];
    v2[i] = v; sum2 += v; sq2 += v * v;
  }
  sum2 += __shfl_xor(sum2, 16); sq2 += __shfl_xor(sq2, 16);
  const float mu2 = sum2 * (1.0f / 64.0f);
  const float vr2 = sq2 * (1.0f / 64.0f) - mu2 * mu2;
  const float ri2 = rsqrtf(vr2 + 1e-5f);
#pragma unroll
  for (int i = 0; i < 32; ++i) {
    const int c = chalf + i;
    s_T[wid][t][c] = (v2[i] - mu2) * ri2 * ln_g[c] + ln_b[c];
  }
  __syncthreads();

  // ---- fused output: x @ weights_pool + res2 @ Wout + bout ----
  bhalf16 ga0 = ld_a16(&s_T[wid][0][0],      66, lane);
  bhalf16 ga1 = ld_a16(&s_T[wid][0][0] + 32, 66, lane);
  bhalf16 xa[6];
#pragma unroll
  for (int ks = 0; ks < 6; ++ks)
    xa[ks] = ld_pk(xpak, tt * 6 + ks, lane);
#pragma unroll
  for (int j = 0; j < 12; ++j) {
    f32x8 c_ = {};
#pragma unroll
    for (int ks = 0; ks < 6; ++ks)
      c_ = wmma_bf16(xa[ks], ld_pk(pw + PW_POOL, ks * 12 + j, lane), c_);
    c_ = wmma_bf16(ga0, ld_pk(pw + PW_OUT, 0 * 12 + j, lane), c_);
    c_ = wmma_bf16(ga1, ld_pk(pw + PW_OUT, 1 * 12 + j, lane), c_);
    const int col = j * 16 + c16;
    const float bb = bout[col];
#pragma unroll
    for (int r = 0; r < 8; ++r)
      out[(size_t)(t0 + r + hi * 8) * OD + col] = c_[r] + bb;
  }
}

// ---------------------------------------------------------------------------
extern "C" void kernel_launch(void* const* d_in, const int* in_sizes, int n_in,
                              void* d_out, int out_size, void* d_ws, size_t ws_size,
                              hipStream_t stream) {
  const float* x     = (const float*)d_in[0];
  const float* Wq    = (const float*)d_in[1];
  const float* bq    = (const float*)d_in[2];
  const float* Wk    = (const float*)d_in[3];
  const float* bk    = (const float*)d_in[4];
  const float* Wv    = (const float*)d_in[5];
  const float* bv    = (const float*)d_in[6];
  const float* Win   = (const float*)d_in[7];
  const float* Wout  = (const float*)d_in[8];
  const float* bout  = (const float*)d_in[9];
  const float* W1    = (const float*)d_in[10];
  const float* b1    = (const float*)d_in[11];
  const float* W2    = (const float*)d_in[12];
  const float* b2    = (const float*)d_in[13];
  const float* ln_g  = (const float*)d_in[14];
  const float* ln_b  = (const float*)d_in[15];
  const float* wpool = (const float*)d_in[16];
  const float* Pi    = (const float*)d_in[17];

  float*  ws   = (float*)d_ws;
  float*  Qws  = ws + OFF_Q;
  float*  Pws  = ws + OFF_P;
  float*  ACC  = ws + OFF_ACC;
  float*  FKVN = ws + OFF_FKVN;
  __bf16* PW   = (__bf16*)(ws + OFF_PAKW);
  __bf16* XPAK = (__bf16*)(ws + OFF_XPAK);
  float*  o    = (float*)d_out;

  // pack weights into bf16 B-fragment layout
  k_packw<<<24, 32, 0, stream>>>(Wq,    TD, PW + PW_Q);
  k_packw<<<24, 32, 0, stream>>>(Wk,    TD, PW + PW_K);
  k_packw<<<24, 32, 0, stream>>>(Wv,    TD, PW + PW_V);
  k_packw<<<24, 32, 0, stream>>>(Win,   TD, PW + PW_IN);
  k_packw<<<16, 32, 0, stream>>>(W1,    FF, PW + PW_1);
  k_packw<<<16, 32, 0, stream>>>(W2,    TD, PW + PW_2);
  k_packw<<<24, 32, 0, stream>>>(Wout,  OD, PW + PW_OUT);
  k_packw<<<72, 32, 0, stream>>>(wpool, OD, PW + PW_POOL);

  k_zero<<<(ACC_N + 255) / 256, 256, 0, stream>>>(ACC, ACC_N);
  k_proj_stats<<<TOK / 128, 256, 0, stream>>>(x, PW, bq, bk, bv,
                                              Qws, Pws, ACC, XPAK);
  k_fin<<<Bc * 4, 256, 0, stream>>>(ACC, Pi, FKVN);
  k_tail<<<TOK / 64, 128, 0, stream>>>(XPAK, Qws, Pws, FKVN, PW,
                                       b1, b2, ln_g, ln_b, bout, o);
}
